// ContextualAttention_76665166233772
// MI455X (gfx1250) — matrix operand order repaired and checked by reading, not compiled, not run on hardware
//
#include <hip/hip_runtime.h>
#include <hip/hip_bf16.h>

// ---------------- problem constants ----------------
#define BB    4
#define CC    128
#define HH    96
#define WW    96
#define HS    48              // downsampled spatial (rate=2)
#define LL    (HS*HS)         // 2304 patch positions
#define KDIM  (CC*9)          // 1152 : C*3*3 matching-patch length
#define MROWS (CC*16)         // 2048 : C*4*4 raw-patch length
#define EPS_SUM 0.1152f       // KDIM * 1e-4 (EPS summed over patch)
#define ATT_SCALE 10.0f

#define BS_STRIDE 48          // padded LDS column stride (elements) -> fewer bank conflicts

typedef __attribute__((ext_vector_type(16))) __bf16 bf16x16;
typedef __attribute__((ext_vector_type(8)))  float  f32x8;
typedef __attribute__((ext_vector_type(4)))  int    i32x4;

union BFrag { uint4 q[2]; bf16x16 v; };

// CDNA5 async global->LDS path (ASYNCcnt), with portable fallback
#if defined(__has_builtin)
#  if __has_builtin(__builtin_amdgcn_global_load_async_to_lds_b128) && \
      __has_builtin(__builtin_amdgcn_s_wait_asynccnt)
#    define USE_ASYNC_LDS 1
#  endif
#endif
#ifndef USE_ASYNC_LDS
#  define USE_ASYNC_LDS 0
#endif

#define AS1 __attribute__((address_space(1)))
#define AS3 __attribute__((address_space(3)))

// ---------------------------------------------------------------
// im2col of downsampled image: P[l][k], l = lh*48+lw, k = c*9+p*3+q
// value = x[c, 2*(lh+p-1), 2*(lw+q-1)] (zero pad), stored bf16.
// ---------------------------------------------------------------
__global__ void pack_P(const float* __restrict__ x, __hip_bfloat16* __restrict__ P) {
    int idx = blockIdx.x * blockDim.x + threadIdx.x;
    if (idx >= LL * KDIM) return;
    int l = idx / KDIM, k = idx - l * KDIM;
    int c = k / 9,      pq = k - c * 9;
    int p = pq / 3,     q  = pq - p * 3;
    int lh = l / HS,    lw = l - lh * HS;
    int y = lh + p - 1, xx = lw + q - 1;
    float v = 0.0f;
    if ((unsigned)y < HS && (unsigned)xx < HS)
        v = x[((size_t)c * HH + 2 * y) * WW + 2 * xx];
    P[idx] = __float2bfloat16(v);
}

// ---------------------------------------------------------------
// Raw 4x4 stride-2 patches, transposed: Rt[c*16+p*4+q][l]
// value = x[c, 2*lh+p-1, 2*lw+q-1] (zero pad), stored bf16.
// ---------------------------------------------------------------
__global__ void pack_R(const float* __restrict__ x, __hip_bfloat16* __restrict__ Rt) {
    int idx = blockIdx.x * blockDim.x + threadIdx.x;
    if (idx >= MROWS * LL) return;
    int row = idx / LL, l = idx - row * LL;
    int c = row >> 4,   pq = row & 15;
    int p = pq >> 2,    q  = pq & 3;
    int lh = l / HS,    lw = l - lh * HS;
    int Y = 2 * lh + p - 1, X = 2 * lw + q - 1;
    float v = 0.0f;
    if ((unsigned)Y < HH && (unsigned)X < WW)
        v = x[((size_t)c * HH + Y) * WW + X];
    Rt[idx] = __float2bfloat16(v);
}

// per-pixel channel sum of squares of downsampled image
__global__ void sumsq_pix(const float* __restrict__ x, float* __restrict__ S) {
    int idx = blockIdx.x * blockDim.x + threadIdx.x;
    if (idx >= HS * HS) return;
    int y = idx / HS, xx = idx - y * HS;
    float s = 0.0f;
    for (int c = 0; c < CC; ++c) {
        float v = x[((size_t)c * HH + 2 * y) * WW + 2 * xx];
        s += v * v;
    }
    S[idx] = s;
}

// per-patch 1/norm and mask flag mf (3x3 mask patch entirely zero -> 1)
__global__ void norm_mf(const float* __restrict__ S, const float* __restrict__ mask,
                        float* __restrict__ invnorm, float* __restrict__ mf) {
    int l = blockIdx.x * blockDim.x + threadIdx.x;
    if (l >= LL) return;
    int lh = l / HS, lw = l - lh * HS;
    float ss = 0.0f, ms = 0.0f;
    for (int dp = 0; dp < 3; ++dp)
        for (int dq = 0; dq < 3; ++dq) {
            int y = lh + dp - 1, xx = lw + dq - 1;
            if ((unsigned)y < HS && (unsigned)xx < HS) {
                ss += S[y * HS + xx];
                ms += mask[(size_t)(2 * y) * WW + 2 * xx];
            }
        }
    invnorm[l] = rsqrtf(ss + EPS_SUM);
    mf[l] = (ms == 0.0f) ? 1.0f : 0.0f;
}

// ---------------------------------------------------------------
// bf16 WMMA GEMM:  D = A * B^T  with A:[Mrows][K], Bmat:[Ncols][K]
// (both row-major, K contiguous).
//   block = 128 threads (4 waves); block tile = 128(M) x 64(N)
//   wave tile = 32(M) x 64(N)  (2x4 = 8 v_wmma_f32_16x16x32_bf16 / K-step)
//   B strip (64x32) staged in LDS, double-buffered, async global->LDS copy
// STORE_ROWMAJOR=1: D[row*LL+col]   (contrib matrix)
// STORE_ROWMAJOR=0: D[col*LL+row]   (Gram stored transposed -> Gt[m][l])
// ---------------------------------------------------------------
template <bool STORE_ROWMAJOR>
__global__ void __launch_bounds__(128) gemm_bf16(
    const __hip_bfloat16* __restrict__ Amat,
    const __hip_bfloat16* __restrict__ Bmat,
    float* __restrict__ D, int K) {
    const int tid  = threadIdx.x;
    const int lane = tid & 31;
    const int wave = tid >> 5;
    const int half = lane >> 4;   // which half of the wave
    const int mr   = lane & 15;   // row (A) / col (B) within tile

    const int n0 = blockIdx.x * 64;                 // N strip base (4 tiles)
    const int r0 = blockIdx.y * 128 + wave * 32;    // this wave's 32 M rows

    __shared__ __align__(16) __hip_bfloat16 Bs[2][64 * BS_STRIDE];

    f32x8 acc[2][4];
#pragma unroll
    for (int m = 0; m < 2; ++m)
#pragma unroll
        for (int t = 0; t < 4; ++t) acc[m][t] = (f32x8)0.0f;

    const char* Ab = (const char*)Amat;
    const char* Bb = (const char*)Bmat;

    // each of the 128 threads stages 32B of the 64x32 B strip
    const int scol = tid >> 1;   // 0..63 : column of strip
    const int sseg = tid & 1;    // 0..1  : k segment (0..15 / 16..31)

    auto stage = [&](int kb, int buf) {
        const char* src = Bb + ((size_t)(n0 + scol) * K + kb + sseg * 16) * 2;
        char* dst = (char*)&Bs[buf][scol * BS_STRIDE + sseg * 16];
#if USE_ASYNC_LDS
        __builtin_amdgcn_global_load_async_to_lds_b128(
            (AS1 i32x4*)src,        (AS3 i32x4*)dst,        0, 0);
        __builtin_amdgcn_global_load_async_to_lds_b128(
            (AS1 i32x4*)(src + 16), (AS3 i32x4*)(dst + 16), 0, 0);
#else
        const uint4* s = (const uint4*)src;
        uint4* d = (uint4*)dst;
        d[0] = s[0];
        d[1] = s[1];
#endif
    };

    stage(0, 0);
#if USE_ASYNC_LDS
    __builtin_amdgcn_s_wait_asynccnt(0);
#endif
    __syncthreads();

    int buf = 0;
    for (int kb = 0; kb < K; kb += 32, buf ^= 1) {
        if (kb + 32 < K) stage(kb + 32, buf ^ 1);  // overlap copy with compute

        // A 16x32 fragments (two M sub-tiles), direct from global (L2 resident)
        BFrag a0, a1;
        const uint4* pa0 = (const uint4*)(Ab + ((size_t)(r0 + mr) * K + kb) * 2);
        a0.q[0] = pa0[half];
        a0.q[1] = pa0[2 + half];
        const uint4* pa1 = (const uint4*)(Ab + ((size_t)(r0 + 16 + mr) * K + kb) * 2);
        a1.q[0] = pa1[half];
        a1.q[1] = pa1[2 + half];

#pragma unroll
        for (int t = 0; t < 4; ++t) {
            // B 32x16 fragment from LDS: lane half h -> K 8h..8h+7, 16+8h..16+8h+7
            BFrag b;
            const uint4* pb =
                (const uint4*)((const char*)&Bs[buf][(t * 16 + mr) * BS_STRIDE]);
            b.q[0] = pb[half];
            b.q[1] = pb[2 + half];
            acc[0][t] = __builtin_amdgcn_wmma_f32_16x16x32_bf16(
                false, a0.v, false, b.v, (short)0, acc[0][t], false, false);
            acc[1][t] = __builtin_amdgcn_wmma_f32_16x16x32_bf16(
                false, a1.v, false, b.v, (short)0, acc[1][t], false, false);
        }

#if USE_ASYNC_LDS
        __builtin_amdgcn_s_wait_asynccnt(0);
#endif
        __syncthreads();
    }

    // C layout: VGPR r holds (M = r + 8*half, N = mr)
#pragma unroll
    for (int m = 0; m < 2; ++m) {
        const int rm = r0 + m * 16;
#pragma unroll
        for (int t = 0; t < 4; ++t) {
            const int col = n0 + t * 16 + mr;
#pragma unroll
            for (int r = 0; r < 8; ++r) {
                const int row = rm + half * 8 + r;
                if (STORE_ROWMAJOR) D[(size_t)row * LL + col] = acc[m][t][r];
                else                D[(size_t)col * LL + row] = acc[m][t][r];
            }
        }
    }
}

// ---------------------------------------------------------------
// softmax over l for each output position m (row of Gt), exactly
// mirroring reference: z = (G[l,m]/norm[l]) * mf[l] * 10, softmax
// over ALL l (masked entries contribute exp(0)), then * mf[l].
// ---------------------------------------------------------------
__global__ void softmax_row(const float* __restrict__ Gt,
                            const float* __restrict__ invnorm,
                            const float* __restrict__ mf,
                            __hip_bfloat16* __restrict__ AttB) {
    const int m = blockIdx.x;
    const int tid = threadIdx.x;
    const float* row = Gt + (size_t)m * LL;
    __shared__ float red[256];

    float mx = -1e30f;
    for (int l = tid; l < LL; l += 256) {
        float z = row[l] * invnorm[l] * mf[l] * ATT_SCALE;
        mx = fmaxf(mx, z);
    }
    red[tid] = mx; __syncthreads();
    for (int s = 128; s > 0; s >>= 1) {
        if (tid < s) red[tid] = fmaxf(red[tid], red[tid + s]);
        __syncthreads();
    }
    const float rowmax = red[0];
    __syncthreads();

    float sum = 0.0f;
    for (int l = tid; l < LL; l += 256) {
        float z = row[l] * invnorm[l] * mf[l] * ATT_SCALE;
        sum += __expf(z - rowmax);
    }
    red[tid] = sum; __syncthreads();
    for (int s = 128; s > 0; s >>= 1) {
        if (tid < s) red[tid] += red[tid + s];
        __syncthreads();
    }
    const float inv = 1.0f / red[0];

    for (int l = tid; l < LL; l += 256) {
        float z = row[l] * invnorm[l] * mf[l] * ATT_SCALE;
        AttB[(size_t)m * LL + l] = __float2bfloat16(__expf(z - rowmax) * inv * mf[l]);
    }
}

// ---------------------------------------------------------------
// overlap-add as a gather: out pixel (c,Y0,X0) <- up to 4 cells of
// Contrib[(c*16+p*4+q)][h*48+w] with Y0+1 = p+2h, X0+1 = q+2w,
// then blend: mask*out + (1-mask)*x.
// ---------------------------------------------------------------
__global__ void gather_blend(const float* __restrict__ Contrib,
                             const float* __restrict__ x,
                             const float* __restrict__ mask,
                             float* __restrict__ out) {
    int idx = blockIdx.x * blockDim.x + threadIdx.x;
    if (idx >= CC * HH * WW) return;
    int c = idx / (HH * WW);
    int rem = idx - c * (HH * WW);
    int Y0 = rem / WW, X0 = rem - Y0 * WW;
    int Y = Y0 + 1, X = X0 + 1;  // crop offset [1:1+H]
    float acc = 0.0f;
    int pbase = Y & 1, qbase = X & 1;
#pragma unroll
    for (int pi = 0; pi < 2; ++pi) {
        int p = pbase + 2 * pi;
        int h = (Y - p) >> 1;
        if ((unsigned)h >= HS) continue;
#pragma unroll
        for (int qi = 0; qi < 2; ++qi) {
            int q = qbase + 2 * qi;
            int w = (X - q) >> 1;
            if ((unsigned)w >= HS) continue;
            acc += Contrib[(size_t)(c * 16 + p * 4 + q) * LL + h * HS + w];
        }
    }
    float mv = mask[(size_t)Y0 * WW + X0];
    out[idx] = mv * acc + (1.0f - mv) * x[idx];
}

// ---------------------------------------------------------------
extern "C" void kernel_launch(void* const* d_in, const int* in_sizes, int n_in,
                              void* d_out, int out_size, void* d_ws, size_t ws_size,
                              hipStream_t stream) {
    (void)in_sizes; (void)n_in; (void)out_size; (void)ws_size;
    const float* x    = (const float*)d_in[0];   // (4,128,96,96)
    const float* mask = (const float*)d_in[1];   // (4,1,96,96)
    float* out = (float*)d_out;                  // (4,128,96,96)

    // workspace carve-out (reused across the per-batch loop); peak ~65.5 MB
    char* ws = (char*)d_ws;
    size_t off = 0;
    auto carve = [&](size_t bytes) {
        void* p = ws + off;
        off += (bytes + 255) & ~(size_t)255;
        return p;
    };
    __hip_bfloat16* P       = (__hip_bfloat16*)carve((size_t)LL * KDIM * 2);
    __hip_bfloat16* Rt      = (__hip_bfloat16*)carve((size_t)MROWS * LL * 2);
    float*          Gt      = (float*)carve((size_t)LL * LL * 4);
    __hip_bfloat16* AttB    = (__hip_bfloat16*)carve((size_t)LL * LL * 2);
    float*          Contrib = (float*)carve((size_t)MROWS * LL * 4);
    float*          S       = (float*)carve((size_t)HS * HS * 4);
    float*          invnorm = (float*)carve((size_t)LL * 4);
    float*          mfbuf   = (float*)carve((size_t)LL * 4);

    for (int b = 0; b < BB; ++b) {
        const float* xb = x    + (size_t)b * CC * HH * WW;
        const float* mb = mask + (size_t)b * HH * WW;
        float*       ob = out  + (size_t)b * CC * HH * WW;

        pack_P<<<(LL * KDIM + 255) / 256, 256, 0, stream>>>(xb, P);
        pack_R<<<(MROWS * LL + 255) / 256, 256, 0, stream>>>(xb, Rt);
        sumsq_pix<<<(HS * HS + 255) / 256, 256, 0, stream>>>(xb, S);
        norm_mf<<<(LL + 255) / 256, 256, 0, stream>>>(S, mb, invnorm, mfbuf);

        // Gram: G = P^T P  (M=N=2304, K=1152) -> stored transposed Gt[m][l]
        gemm_bf16<false><<<dim3(LL / 64, LL / 128), 128, 0, stream>>>(P, P, Gt, KDIM);

        softmax_row<<<LL, 256, 0, stream>>>(Gt, invnorm, mfbuf, AttB);

        // contrib = R^T * Att  (M=2048, N=2304, K=2304)
        gemm_bf16<true><<<dim3(LL / 64, MROWS / 128), 128, 0, stream>>>(Rt, AttB, Contrib, LL);

        gather_blend<<<(CC * HH * WW + 255) / 256, 256, 0, stream>>>(Contrib, xb, mb, ob);
    }
}
